// edge_update_20976620274310
// MI455X (gfx1250) — compile-verified, matrix operand order checked by reading, not run on hardware
//
#include <hip/hip_runtime.h>
#include <hip/hip_bf16.h>

// CDNA5 / gfx1250. Wave32. WMMA f16 (f32 accum) for both GEMMs.
typedef __attribute__((ext_vector_type(16))) _Float16 v16h;
typedef __attribute__((ext_vector_type(8)))  float    v8f;

#define NODE_DIM 128
#define EDGE_DIM 128
#define CAT_DIM  384
#define LN_EPS   1e-5f
#define SLOPE    0.01f

#define WAVES_PER_BLOCK 8
#define EDGES_PER_WAVE  16
#define EDGES_PER_BLOCK (WAVES_PER_BLOCK * EDGES_PER_WAVE)   // 128

#define KT1 12               // 384 / 32
#define KT2 4                // 128 / 32
#define NT  8                // 128 / 16
#define FRAG_HALFS 512       // 32 lanes * 16 halves per 32x16 B-fragment
#define W1_HALFS (KT1 * NT * FRAG_HALFS)   // 49152 halves = 96 KB
#define W2_HALFS (KT2 * NT * FRAG_HALFS)   // 16384 halves = 32 KB

// ---------------------------------------------------------------------------
// Prep: convert f32 weights [K, N] into f16 WMMA B-fragments.
// Fragment (kt, nt): lane L holds column n = nt*16 + (L&15); half h holds
// k = kt*32 + (L>>4)*8 + (h>>3)*16 + (h&7)   (16-bit operand layout, ISA 7.12.2)
// ---------------------------------------------------------------------------
__global__ void prep_weights_kernel(const float* __restrict__ W1,
                                    const float* __restrict__ W2,
                                    _Float16* __restrict__ wf) {
    int t = blockIdx.x * blockDim.x + threadIdx.x;
    if (t >= W1_HALFS + W2_HALFS) return;
    const float* W = W1;
    int tt = t;
    if (t >= W1_HALFS) { W = W2; tt = t - W1_HALFS; }
    int f      = tt >> 9;        // fragment index = kt*8 + nt
    int within = tt & 511;
    int lane   = within >> 4;
    int h      = within & 15;
    int kt = f >> 3, nt = f & 7;
    int n = nt * 16 + (lane & 15);
    int k = kt * 32 + (lane >> 4) * 8 + (h >> 3) * 16 + (h & 7);
    wf[t] = (_Float16)W[k * EDGE_DIM + n];
}

// ---------------------------------------------------------------------------
// Main fused kernel: gather -> LayerNorm -> GEMM1 -> LeakyReLU -> GEMM2 -> +e
// ---------------------------------------------------------------------------
__global__ __launch_bounds__(256)
void edge_update_main_kernel(const float* __restrict__ node_feat,
                             const int*   __restrict__ edge_src,
                             const int*   __restrict__ edge_dst,
                             const float* __restrict__ e,
                             const float* __restrict__ gamma,
                             const float* __restrict__ beta,
                             const float* __restrict__ b1,
                             const float* __restrict__ b2,
                             const _Float16* __restrict__ wf,
                             float* __restrict__ out,
                             int nEdges) {
    __shared__ alignas(32) _Float16 w1_lds[W1_HALFS];                    // 96 KB
    __shared__ float g_lds[CAT_DIM];
    __shared__ float bt_lds[CAT_DIM];
    __shared__ alignas(16) _Float16 h1_lds[WAVES_PER_BLOCK * 16 * EDGE_DIM]; // 32 KB

    const int tid = threadIdx.x;

    // Stage W1 fragments into LDS (uint4 = 8 halves per copy).
    {
        const uint4* s = (const uint4*)wf;
        uint4*       d = (uint4*)w1_lds;
        for (int i = tid; i < W1_HALFS / 8; i += 256) d[i] = s[i];
        for (int i = tid; i < CAT_DIM; i += 256) {
            g_lds[i]  = gamma[i];
            bt_lds[i] = beta[i];
        }
    }
    __syncthreads();

    const int lane   = tid & 31;
    const int wave   = tid >> 5;
    const int r      = lane & 15;    // row within 16-edge tile (A/C row or C col)
    const int halfid = lane >> 4;    // which half of the wave

    const int e0   = blockIdx.x * EDGES_PER_BLOCK + wave * EDGES_PER_WAVE;
    int eidx = e0 + r;
    if (eidx >= nEdges) eidx = nEdges - 1;   // clamp loads; stores guarded below

    const float* xi = node_feat + (size_t)edge_src[eidx] * NODE_DIM;
    const float* xj = node_feat + (size_t)edge_dst[eidx] * NODE_DIM;
    const float* er = e + (size_t)eidx * EDGE_DIM;

    // cat[r][c] fetch, c aligned to 4 and never crossing a 128-boundary.
    auto cat4 = [&](int c) -> float4 {
        const float* p = (c < NODE_DIM) ? (xi + c)
                       : (c < 2 * NODE_DIM) ? (xj + (c - NODE_DIM))
                       : (er + (c - 2 * NODE_DIM));
        return *(const float4*)p;
    };

    // ---- Pass 1: LayerNorm statistics (each lane covers 192 columns) ----
    float s = 0.f, ss = 0.f;
    const int cbase = halfid * 192;
#pragma unroll 4
    for (int i = 0; i < 48; ++i) {
        float4 v = cat4(cbase + i * 4);
        s  += v.x + v.y + v.z + v.w;
        ss += v.x * v.x + v.y * v.y + v.z * v.z + v.w * v.w;
    }
    s  += __shfl_xor(s, 16, 32);
    ss += __shfl_xor(ss, 16, 32);
    const float mu   = s * (1.f / CAT_DIM);
    const float rstd = rsqrtf(ss * (1.f / CAT_DIM) - mu * mu + LN_EPS);

    // ---- GEMM1: z(16x384) @ W1(384x128), K in 12 tiles of 32 ----
    v8f acc[NT];
#pragma unroll
    for (int i = 0; i < NT; ++i) acc[i] = (v8f){0.f, 0.f, 0.f, 0.f, 0.f, 0.f, 0.f, 0.f};

    for (int kt = 0; kt < KT1; ++kt) {
        union { v16h v; _Float16 h[16]; } af;
        const int k0 = kt * 32 + halfid * 8;
#pragma unroll
        for (int seg = 0; seg < 2; ++seg) {
            const int kb = k0 + seg * 16;
            float4 a0 = cat4(kb);
            float4 a1 = cat4(kb + 4);
            float va[8] = {a0.x, a0.y, a0.z, a0.w, a1.x, a1.y, a1.z, a1.w};
#pragma unroll
            for (int j = 0; j < 8; ++j) {
                const int k = kb + j;
                const float z = (va[j] - mu) * rstd * g_lds[k] + bt_lds[k];
                af.h[seg * 8 + j] = (_Float16)z;
            }
        }
        const _Float16* wbase = &w1_lds[(kt * NT) * FRAG_HALFS + lane * 16];
#pragma unroll
        for (int nt = 0; nt < NT; ++nt) {
            v16h bf = *(const v16h*)(wbase + nt * FRAG_HALFS);
            acc[nt] = __builtin_amdgcn_wmma_f32_16x16x32_f16(
                false, af.v, false, bf, (short)0, acc[nt], false, false);
        }
    }

    // ---- bias + LeakyReLU, transpose C-layout -> A-layout through LDS ----
    _Float16* h1w = &h1_lds[wave * 16 * EDGE_DIM];
#pragma unroll
    for (int nt = 0; nt < NT; ++nt) {
        const int n = nt * 16 + r;            // C/D: lane gives column
        const float b1n = b1[n];
#pragma unroll
        for (int v = 0; v < 8; ++v) {
            const int m = halfid * 8 + v;     // C/D: VGPR v + 8*(lane>>4) gives row
            float x = acc[nt][v] + b1n;
            x = (x >= 0.f) ? x : SLOPE * x;
            h1w[m * EDGE_DIM + n] = (_Float16)x;
        }
    }
    __syncthreads();

    // ---- GEMM2: h1(16x128) @ W2(128x128), K in 4 tiles of 32 ----
    v8f acc2[NT];
#pragma unroll
    for (int i = 0; i < NT; ++i) acc2[i] = (v8f){0.f, 0.f, 0.f, 0.f, 0.f, 0.f, 0.f, 0.f};

    const _Float16* w2g = wf + W1_HALFS;      // W2 fragments stay in L2
    for (int kt = 0; kt < KT2; ++kt) {
        union { v16h v; uint4 u[2]; } af;
        const int k0 = kt * 32 + halfid * 8;
        af.u[0] = *(const uint4*)&h1w[r * EDGE_DIM + k0];
        af.u[1] = *(const uint4*)&h1w[r * EDGE_DIM + k0 + 16];
#pragma unroll
        for (int nt = 0; nt < NT; ++nt) {
            v16h bf = *(const v16h*)(w2g + (kt * NT + nt) * FRAG_HALFS + lane * 16);
            acc2[nt] = __builtin_amdgcn_wmma_f32_16x16x32_f16(
                false, af.v, false, bf, (short)0, acc2[nt], false, false);
        }
    }

    // ---- epilogue: out = e + h2 + b2 ----
#pragma unroll
    for (int nt = 0; nt < NT; ++nt) {
        const int n = nt * 16 + r;
        const float b2n = b2[n];
#pragma unroll
        for (int v = 0; v < 8; ++v) {
            const int m = halfid * 8 + v;
            const int eo = e0 + m;
            if (eo < nEdges) {
                out[(size_t)eo * EDGE_DIM + n] =
                    e[(size_t)eo * EDGE_DIM + n] + acc2[nt][v] + b2n;
            }
        }
    }
}

// ---------------------------------------------------------------------------
extern "C" void kernel_launch(void* const* d_in, const int* in_sizes, int n_in,
                              void* d_out, int out_size, void* d_ws, size_t ws_size,
                              hipStream_t stream) {
    const float* node_feat = (const float*)d_in[0];
    const int*   edge_src  = (const int*)d_in[1];
    const int*   edge_dst  = (const int*)d_in[2];
    const float* e         = (const float*)d_in[3];
    const float* ln_gamma  = (const float*)d_in[4];
    const float* ln_beta   = (const float*)d_in[5];
    const float* W1        = (const float*)d_in[6];
    const float* b1        = (const float*)d_in[7];
    const float* W2        = (const float*)d_in[8];
    const float* b2        = (const float*)d_in[9];
    float* out = (float*)d_out;

    const int nEdges = in_sizes[3] / EDGE_DIM;
    _Float16* wf = (_Float16*)d_ws;   // 128 KB of f16 weight fragments

    const int prepTotal = W1_HALFS + W2_HALFS;
    prep_weights_kernel<<<(prepTotal + 255) / 256, 256, 0, stream>>>(W1, W2, wf);

    const int nBlocks = (nEdges + EDGES_PER_BLOCK - 1) / EDGES_PER_BLOCK;
    edge_update_main_kernel<<<nBlocks, 256, 0, stream>>>(
        node_feat, edge_src, edge_dst, e, ln_gamma, ln_beta, b1, b2, wf, out, nEdges);
}
